// HypAgg_16260746182751
// MI455X (gfx1250) — matrix operand order repaired and checked by reading, not compiled, not run on hardware
//
#include <hip/hip_runtime.h>
#include <math.h>

typedef __attribute__((ext_vector_type(16))) _Float16 v16h;
typedef __attribute__((ext_vector_type(8)))  float    v8f;

#define WPB 8  // waves per block (256 threads, wave32)

__device__ __forceinline__ float wave_sum(float v) {
#pragma unroll
    for (int o = 16; o > 0; o >>= 1) v += __shfl_xor(v, o, 32);
    return v;
}

// ---------------- degree counts ----------------
__global__ void count_kernel(const long long* __restrict__ adj,
                             int* __restrict__ cnt_s, int* __restrict__ cnt_d, int E) {
    int t = blockIdx.x * blockDim.x + threadIdx.x;
    if (t < E) {
        atomicAdd(&cnt_s[(int)adj[t]], 1);
        atomicAdd(&cnt_d[(int)adj[(size_t)t + (size_t)E]], 1);
    }
}

// ---------------- logmap0 + per-node attention partials ----------------
// one wave per node; xt written into d_out (used as scratch until final pass)
__global__ void node_kernel(const float* __restrict__ x, const float* __restrict__ attw,
                            float* __restrict__ xt_out,
                            float* __restrict__ s0, float* __restrict__ s1, int N) {
    int wid  = (blockIdx.x * blockDim.x + threadIdx.x) >> 5;
    int lane = threadIdx.x & 31;
    if (wid >= N) return;
    float4 v = ((const float4*)(x + (size_t)wid * 128))[lane];
    float nsq = wave_sum(v.x * v.x + v.y * v.y + v.z * v.z + v.w * v.w);
    float pn = fmaxf(sqrtf(nsq), 1e-15f);
    float y  = fminf(pn, 1.0f - 1e-7f);
    float sc = atanhf(y) / pn;
    float4 t;
    t.x = v.x * sc; t.y = v.y * sc; t.z = v.z * sc; t.w = v.w * sc;
    ((float4*)(xt_out + (size_t)wid * 128))[lane] = t;
    float4 a0 = ((const float4*)attw)[lane];
    float4 a1 = ((const float4*)(attw + 128))[lane];
    float p0 = wave_sum(t.x * a0.x + t.y * a0.y + t.z * a0.z + t.w * a0.w);
    float p1 = wave_sum(t.x * a1.x + t.y * a1.y + t.z * a1.z + t.w * a1.w);
    if (lane == 0) { s0[wid] = p0; s1[wid] = p1; }
}

// ---------------- per-edge: WMMA attention dot + atomic scatters ----------------
__global__ void __launch_bounds__(256) edge_kernel(
    const long long* __restrict__ adj, const float* __restrict__ ea,
    const float* __restrict__ attw, const float* __restrict__ attb,
    const float* __restrict__ s0, const float* __restrict__ s1,
    const int* __restrict__ cnt_s, const int* __restrict__ cnt_d,
    float* __restrict__ A, float* __restrict__ att_sum,
    int E, int numTiles, int totalWaves) {
    __shared__ float lds[WPB][16][132];  // +4 pad: kills row-major bank conflicts
    int lane  = threadIdx.x & 31;
    int winb  = threadIdx.x >> 5;
    int gw    = blockIdx.x * WPB + winb;
    float (*T)[132] = lds[winb];
    const long long* srcp = adj;
    const long long* dstp = adj + (size_t)E;
    const float* w2 = attw + 256;
    float bias = attb[0];
    int half8 = (lane & 16) ? 8 : 0;

    // B matrix: w2 chunk replicated across all 16 columns (K layout per ISA:
    // lanes 0-15 hold K=0..7,16..23, lanes 16-31 hold K=8..15,24..31, 2 halves per VGPR)
    v16h bv[4];
#pragma unroll
    for (int w = 0; w < 4; ++w) {
        int base = 32 * w + ((lane & 16) ? 8 : 0);
#pragma unroll
        for (int k = 0; k < 16; ++k) bv[w][k] = (_Float16)w2[base + k + ((k >= 8) ? 8 : 0)];
    }

    for (int tile = gw; tile < numTiles; tile += totalWaves) {
        int e0 = tile * 16;
        {   // prefetch next tile's edge_attr (8 KB, 256B/lane stride)
            int nt = tile + totalWaves;
            if (nt < numTiles)
                __builtin_prefetch(ea + (size_t)nt * 16 * 128 + lane * 64, 0, 1);
        }
        // stage 16 edge rows (coalesced 512B per row)
#pragma unroll 4
        for (int r = 0; r < 16; ++r) {
            int e = e0 + r;
            float4 v = (e < E) ? ((const float4*)(ea + (size_t)e * 128))[lane]
                               : make_float4(0.f, 0.f, 0.f, 0.f);
            *((float4*)&T[r][lane * 4]) = v;
        }
        __builtin_amdgcn_wave_barrier();

        // D = A(16 edges x 128) * w2  via 4x v_wmma_f32_16x16x32_f16
        v8f c = {};
        int row = lane & 15;
#pragma unroll
        for (int w = 0; w < 4; ++w) {
            int kb = 32 * w;
            float4 f0 = *((const float4*)&T[row][kb + half8]);
            float4 f1 = *((const float4*)&T[row][kb + half8 + 4]);
            float4 f2 = *((const float4*)&T[row][kb + 16 + half8]);
            float4 f3 = *((const float4*)&T[row][kb + 16 + half8 + 4]);
            v16h a;
            a[0]=(_Float16)f0.x; a[1]=(_Float16)f0.y; a[2]=(_Float16)f0.z; a[3]=(_Float16)f0.w;
            a[4]=(_Float16)f1.x; a[5]=(_Float16)f1.y; a[6]=(_Float16)f1.z; a[7]=(_Float16)f1.w;
            a[8]=(_Float16)f2.x; a[9]=(_Float16)f2.y; a[10]=(_Float16)f2.z; a[11]=(_Float16)f2.w;
            a[12]=(_Float16)f3.x; a[13]=(_Float16)f3.y; a[14]=(_Float16)f3.z; a[15]=(_Float16)f3.w;
            c = __builtin_amdgcn_wmma_f32_16x16x32_f16(false, a, false, bv[w],
                                                       (short)0, c, false, false);
        }

        // lane j (j<16) owns edge e0+j metadata
        int  eL = e0 + row;
        bool vL = eL < E;
        int  sL = vL ? (int)srcp[eL] : 0;
        int  dL = vL ? (int)dstp[eL] : 0;
        float sv = vL ? (s0[sL] + s1[dL]) : 0.f;

        // attention sigmoid + scalar att_sum atomic (edge i in c[i] for lanes 0-15,
        // edge i+8 in c[i] for lanes 16-31; all columns identical)
#pragma unroll
        for (int i = 0; i < 8; ++i) {
            int   tgt  = i + half8;
            float svv  = __shfl(sv, tgt, 32);
            int   ss   = __shfl(sL, tgt, 32);
            float logit = c[i] + svv + bias;
            float att   = 1.0f / (1.0f + expf(-logit));
            if ((lane == 0 || lane == 16) && (e0 + tgt < E))
                unsafeAtomicAdd(att_sum + ss, att);
        }

        // scatter edge_attr/deg into shared accumulator A (f32 atomics hit L2)
        for (int i = 0; i < 16; ++i) {
            int e = e0 + i;
            if (e >= E) break;
            int ss = __shfl(sL, i, 32);
            int dd = __shfl(dL, i, 32);
            int cs = cnt_s[ss]; if (cs < 1) cs = 1;
            int cd = cnt_d[dd]; if (cd < 1) cd = 1;
            float ics = 1.0f / (float)cs;
            float icd = 1.0f / (float)cd;
            float4 v = *((const float4*)&T[i][lane * 4]);
            float* As = A + (size_t)ss * 128 + lane * 4;
            float* Ad = A + (size_t)dd * 128 + lane * 4;
            unsafeAtomicAdd(As + 0, v.x * ics); unsafeAtomicAdd(As + 1, v.y * ics);
            unsafeAtomicAdd(As + 2, v.z * ics); unsafeAtomicAdd(As + 3, v.w * ics);
            unsafeAtomicAdd(Ad + 0, v.x * icd); unsafeAtomicAdd(Ad + 1, v.y * icd);
            unsafeAtomicAdd(Ad + 2, v.z * icd); unsafeAtomicAdd(Ad + 3, v.w * icd);
        }
        __builtin_amdgcn_wave_barrier();
    }
}

// ---------------- support_t -> expmap0 -> proj ----------------
__global__ void final_kernel(float* __restrict__ xt_io, const float* __restrict__ A,
                             const float* __restrict__ att_sum, int N) {
    int wid  = (blockIdx.x * blockDim.x + threadIdx.x) >> 5;
    int lane = threadIdx.x & 31;
    if (wid >= N) return;
    float4 xt = ((const float4*)(xt_io + (size_t)wid * 128))[lane];
    float4 av = ((const float4*)(A + (size_t)wid * 128))[lane];
    float  as = 1.0f + att_sum[wid];   // xt + xt*att_sum
    float4 u;
    u.x = xt.x * as + av.x; u.y = xt.y * as + av.y;
    u.z = xt.z * as + av.z; u.w = xt.w * as + av.w;
    float un = fmaxf(sqrtf(wave_sum(u.x * u.x + u.y * u.y + u.z * u.z + u.w * u.w)), 1e-15f);
    float th = tanhf(un);
    float sc = th / un;                          // expmap0 scale
    float yn = fmaxf(th, 1e-15f);                // ||expmap0(u)|| == tanh(un)
    const float maxnorm = 1.0f - 4e-3f;
    if (yn > maxnorm) sc *= maxnorm / yn;        // proj
    float4 o;
    o.x = u.x * sc; o.y = u.y * sc; o.z = u.z * sc; o.w = u.w * sc;
    ((float4*)(xt_io + (size_t)wid * 128))[lane] = o;
}

extern "C" void kernel_launch(void* const* d_in, const int* in_sizes, int n_in,
                              void* d_out, int out_size, void* d_ws, size_t ws_size,
                              hipStream_t stream) {
    const float*     x    = (const float*)d_in[0];
    const long long* adj  = (const long long*)d_in[1];
    const float*     ea   = (const float*)d_in[2];
    const float*     attw = (const float*)d_in[3];
    const float*     attb = (const float*)d_in[4];
    float* out = (float*)d_out;

    int N = in_sizes[0] / 128;
    int E = in_sizes[1] / 2;

    // workspace layout: [A: N*128 f][att_sum: N f][cnt_s: N i][cnt_d: N i][s0: N f][s1: N f]
    float* A       = (float*)d_ws;
    float* att_sum = A + (size_t)N * 128;
    int*   cnt_s   = (int*)(att_sum + N);
    int*   cnt_d   = cnt_s + N;
    float* s0      = (float*)(cnt_d + N);
    float* s1      = s0 + N;

    // zero A + att_sum + both count arrays (contiguous)
    hipMemsetAsync(A, 0, (size_t)N * (128 + 3) * sizeof(float), stream);

    count_kernel<<<(E + 255) / 256, 256, 0, stream>>>(adj, cnt_s, cnt_d, E);
    node_kernel<<<(N + 7) / 8, 256, 0, stream>>>(x, attw, out, s0, s1, N);

    int numTiles   = (E + 15) / 16;
    int blocks     = 512;
    int totalWaves = blocks * WPB;
    edge_kernel<<<blocks, 256, 0, stream>>>(adj, ea, attw, attb, s0, s1,
                                            cnt_s, cnt_d, A, att_sum,
                                            E, numTiles, totalWaves);

    final_kernel<<<(N + 7) / 8, 256, 0, stream>>>(out, A, att_sum, N);
}
